// Generator_60026462929085
// MI455X (gfx1250) — matrix operand order repaired and checked by reading, not compiled, not run on hardware
//
#include <hip/hip_runtime.h>
#include <hip/hip_bf16.h>
#include <math.h>

// ---------------------------------------------------------------------------
// Gumbel-softmax LSTM generator for MI455X (gfx1250, wave32, WMMA bf16).
// All GEMMs use v_wmma_f32_16x16x32_bf16 with fragment layouts per
// CDNA5 ISA 7.12.2. Weights converted to bf16 once per call
// (L2-resident: ~30MB of 192MB). Recurrent state: h in bf16 (ping-pong),
// c in fp32. Deterministic counter-based gumbel noise; no atomics.
// Register blocking: 2 M-tiles x 4 N-tiles per wave in the big GEMMs
// (~22 flop/L2-byte vs ~13 unblocked).
// ---------------------------------------------------------------------------

typedef __bf16 bf16_t;
typedef __attribute__((ext_vector_type(16))) __bf16 v16bf;
typedef __attribute__((ext_vector_type(8)))  __bf16 v8bf;
typedef __attribute__((ext_vector_type(8)))  float  v8f;

#define V_  8192
#define E_  256
#define H_  1024
#define T_  128
#define B_  128
#define H4_ 4096

// ---------------- WMMA fragment loaders (ISA 7.12.2 layouts) ----------------

// A: 16x32 bf16 (MxK). lane L: row m = L&15; K = 8*(L>>4) + {0..7} and +16..+23.
__device__ __forceinline__ v16bf load_a_frag(const bf16_t* __restrict__ base,
                                             int ld, int m0, int k0, int lane) {
  const int m  = m0 + (lane & 15);
  const int kb = k0 + 8 * (lane >> 4);
  const bf16_t* p = base + (size_t)m * ld + kb;
  v8bf lo = *(const v8bf*)p;
  v8bf hi = *(const v8bf*)(p + 16);
  v16bf a;
#pragma unroll
  for (int i = 0; i < 8; ++i) { a[i] = lo[i]; a[i + 8] = hi[i]; }
  return a;
}

// B: 32x16 bf16 (KxN), sourced from row-major W (N x K): B[k][n] = W[n][k].
// lane L: column n = L&15; K = 8*(L>>4) + {0..7} and +16..+23.
__device__ __forceinline__ v16bf load_b_frag(const bf16_t* __restrict__ base,
                                             int ld, int n0, int k0, int lane) {
  const int n  = n0 + (lane & 15);
  const int kb = k0 + 8 * (lane >> 4);
  const bf16_t* p = base + (size_t)n * ld + kb;
  v8bf lo = *(const v8bf*)p;
  v8bf hi = *(const v8bf*)(p + 16);
  v16bf b;
#pragma unroll
  for (int i = 0; i < 8; ++i) { b[i] = lo[i]; b[i + 8] = hi[i]; }
  return b;
}

__device__ __forceinline__ v8f wmma_bf16(v16bf a, v16bf b, v8f c) {
  return __builtin_amdgcn_wmma_f32_16x16x32_bf16(false, a, false, b,
                                                 (short)0, c, false, false);
}

__device__ __forceinline__ v8f vzero8() {
  v8f z = {0.f, 0.f, 0.f, 0.f, 0.f, 0.f, 0.f, 0.f};
  return z;
}

// ---------------- deterministic counter RNG (splitmix64) --------------------

__device__ __forceinline__ float u01_hash(unsigned long long x) {
  x ^= x >> 30; x *= 0xbf58476d1ce4e5b9ULL;
  x ^= x >> 27; x *= 0x94d049bb133111ebULL;
  x ^= x >> 31;
  return ((float)(unsigned)(x >> 40) + 0.5f) * (1.0f / 16777216.0f); // (0,1)
}

__device__ __forceinline__ float sigmoidf_(float x) {
  return 1.0f / (1.0f + expf(-x));
}

// ---------------- one-time preprocessing kernels ----------------------------

__global__ void cvt_bf16_kernel(const float* __restrict__ src,
                                bf16_t* __restrict__ dst, int n) {
  for (int i = blockIdx.x * blockDim.x + threadIdx.x; i < n;
       i += gridDim.x * blockDim.x)
    dst[i] = (bf16_t)src[i];
}

// emb (V x E) fp32 -> embT (E x V) bf16, so B-frags of the soft_emb GEMM are
// contiguous per lane.
__global__ void embT_kernel(const float* __restrict__ emb,
                            bf16_t* __restrict__ embT) {
  int idx = blockIdx.x * blockDim.x + threadIdx.x; // over V_*E_
  if (idx >= V_ * E_) return;
  int v = idx >> 8;        // /E_
  int e = idx & (E_ - 1);
  embT[(size_t)e * V_ + v] = (bf16_t)emb[idx];
}

__global__ void bias_kernel(const float* __restrict__ b_ih,
                            const float* __restrict__ b_hh,
                            float* __restrict__ bias) {
  int i = blockIdx.x * blockDim.x + threadIdx.x;
  if (i < H4_) bias[i] = b_ih[i] + b_hh[i];
}

// st = noise @ W_init^T + b_init ; h0 = st[:, :H] (bf16), c0 = st[:, H:]
__global__ void init_state_kernel(const float* __restrict__ noise,
                                  const float* __restrict__ W_init,
                                  const float* __restrict__ b_init,
                                  bf16_t* __restrict__ h_bf,
                                  float* __restrict__ c) {
  int idx = blockIdx.x * blockDim.x + threadIdx.x; // B_*2H
  if (idx >= B_ * 2 * H_) return;
  int b = idx >> 11;            // /2048
  int n = idx & (2 * H_ - 1);
  const float* w = W_init + (size_t)n * 128;
  const float* x = noise + (size_t)b * 128;
  float s = b_init[n];
#pragma unroll 8
  for (int k = 0; k < 128; ++k) s = fmaf(x[k], w[k], s);
  if (n < H_) h_bf[b * H_ + n] = (bf16_t)s;
  else        c[b * H_ + (n - H_)] = s;
}

// ---------------- LSTM cell: z = x@Wih^T + h@Whh^T + bias; gates ------------
// grid: 32 blocks x 256 thr = 256 waves = 4 M-pair-tiles x 64 Jtiles.
// Each wave owns two 16-row M-tiles of the (16-col) j-tile of all four gates;
// the B fragment of each gate is reused across both M-tiles (2x W_hh reuse).

__global__ __launch_bounds__(256) void lstm_cell_kernel(
    const bf16_t* __restrict__ h_in,   // B x H bf16
    const bf16_t* __restrict__ x_in,   // B x E bf16, or nullptr (x == 0)
    float* __restrict__ c,             // B x H fp32 (in/out, tile-owned)
    bf16_t* __restrict__ h_out,        // B x H bf16 (ping-pong)
    const bf16_t* __restrict__ Whh,    // 4H x H bf16
    const bf16_t* __restrict__ Wih,    // 4H x E bf16
    const float* __restrict__ bias) {  // 4H fp32 (b_ih + b_hh)
  const int lane = threadIdx.x & 31;
  const int wave = blockIdx.x * 8 + (threadIdx.x >> 5);
  const int mg = wave >> 6;      // 0..3 (pair of M-tiles)
  const int jt = wave & 63;      // 0..63
  const int m0 = mg * 32;
  const int j0 = jt * 16;

  v8f acc[2][4];
#pragma unroll
  for (int mi = 0; mi < 2; ++mi)
#pragma unroll
    for (int g = 0; g < 4; ++g) acc[mi][g] = vzero8();

  for (int k0 = 0; k0 < H_; k0 += 32) {
    v16bf a0 = load_a_frag(h_in, H_, m0, k0, lane);
    v16bf a1 = load_a_frag(h_in, H_, m0 + 16, k0, lane);
#pragma unroll
    for (int g = 0; g < 4; ++g) {
      v16bf bm = load_b_frag(Whh, H_, g * H_ + j0, k0, lane);
      acc[0][g] = wmma_bf16(a0, bm, acc[0][g]);
      acc[1][g] = wmma_bf16(a1, bm, acc[1][g]);
    }
  }
  if (x_in) { // uniform branch: EXEC stays all-ones for the WMMAs
    for (int k0 = 0; k0 < E_; k0 += 32) {
      v16bf a0 = load_a_frag(x_in, E_, m0, k0, lane);
      v16bf a1 = load_a_frag(x_in, E_, m0 + 16, k0, lane);
#pragma unroll
      for (int g = 0; g < 4; ++g) {
        v16bf bm = load_b_frag(Wih, E_, g * H_ + j0, k0, lane);
        acc[0][g] = wmma_bf16(a0, bm, acc[0][g]);
        acc[1][g] = wmma_bf16(a1, bm, acc[1][g]);
      }
    }
  }

  const int n = j0 + (lane & 15);
  const float bi  = bias[n];
  const float bf_ = bias[H_ + n];
  const float bg  = bias[2 * H_ + n];
  const float bo  = bias[3 * H_ + n];
#pragma unroll
  for (int mi = 0; mi < 2; ++mi) {
    const int mb = m0 + mi * 16 + 8 * (lane >> 4);
#pragma unroll
    for (int r = 0; r < 8; ++r) {
      const int m = mb + r;
      float zi = acc[mi][0][r] + bi;
      float zf = acc[mi][1][r] + bf_;
      float zg = acc[mi][2][r] + bg;
      float zo = acc[mi][3][r] + bo;
      float co = c[m * H_ + n];
      float cn = sigmoidf_(zf) * co + sigmoidf_(zi) * tanhf(zg);
      float hn = sigmoidf_(zo) * tanhf(cn);
      c[m * H_ + n] = cn;
      h_out[m * H_ + n] = (bf16_t)hn;
    }
  }
}

// ---------------- logits + gumbel: s = (h@Wout^T + b_out + g)/tau -----------
// grid: 64 blocks x 256 thr = 512 waves = 4 M-pair-tiles x 128 groups of
// 4 N-tiles. B fragments (W_out rows) reused across both M-tiles.

__global__ __launch_bounds__(256) void logits_gumbel_kernel(
    const bf16_t* __restrict__ h_in,   // B x H bf16
    const bf16_t* __restrict__ Wout,   // V x H bf16
    const float* __restrict__ b_out,   // V
    const float* __restrict__ tau_p,   // scalar
    float* __restrict__ s_out,         // B x V fp32
    unsigned int seed) {
  const int lane = threadIdx.x & 31;
  const int wave = blockIdx.x * 8 + (threadIdx.x >> 5);
  const int mg = wave >> 7;       // 0..3 (pair of M-tiles)
  const int ng = wave & 127;      // 0..127
  const int m0 = mg * 32;
  const int n0 = ng * 64;

  v8f acc[2][4];
#pragma unroll
  for (int mi = 0; mi < 2; ++mi)
#pragma unroll
    for (int q = 0; q < 4; ++q) acc[mi][q] = vzero8();

  for (int k0 = 0; k0 < H_; k0 += 32) {
    v16bf a0 = load_a_frag(h_in, H_, m0, k0, lane);
    v16bf a1 = load_a_frag(h_in, H_, m0 + 16, k0, lane);
#pragma unroll
    for (int q = 0; q < 4; ++q) {
      v16bf bm = load_b_frag(Wout, H_, n0 + q * 16, k0, lane);
      acc[0][q] = wmma_bf16(a0, bm, acc[0][q]);
      acc[1][q] = wmma_bf16(a1, bm, acc[1][q]);
    }
  }

  const float inv_tau = 1.0f / tau_p[0];
#pragma unroll
  for (int q = 0; q < 4; ++q) {
    const int n = n0 + q * 16 + (lane & 15);
    const float bn = b_out[n];
#pragma unroll
    for (int mi = 0; mi < 2; ++mi) {
      const int mb = m0 + mi * 16 + 8 * (lane >> 4);
#pragma unroll
      for (int r = 0; r < 8; ++r) {
        const int m = mb + r;
        float y = acc[mi][q][r] + bn;
        unsigned long long ctr =
            ((unsigned long long)seed << 32) | (unsigned)(m * V_ + n);
        float u = u01_hash(ctr);
        float g = -logf(-logf(u));             // gumbel(0,1)
        s_out[(size_t)m * V_ + n] = (y + g) * inv_tau;
      }
    }
  }
}

// ---------------- softmax over V per row; fixed-order => deterministic ------
// one block per batch row; LDS-staged (32KB) to read s once.

__global__ __launch_bounds__(256) void softmax_kernel(
    const float* __restrict__ s,      // B x V
    bf16_t* __restrict__ w_bf,        // B x V bf16 probs (nullable)
    float* __restrict__ out,          // B x T x V fp32 (nullable)
    int t) {
  __shared__ float sh[V_];
  __shared__ float red[16];
  const int row  = blockIdx.x;
  const int tid  = threadIdx.x;
  const int lane = tid & 31;
  const int wid  = tid >> 5;
  const float* sr = s + (size_t)row * V_;

  float mx = -3.4e38f;
  for (int i = tid; i < V_; i += 256) {
    float v = sr[i];
    sh[i] = v;
    mx = fmaxf(mx, v);
  }
#pragma unroll
  for (int o = 16; o > 0; o >>= 1) mx = fmaxf(mx, __shfl_down(mx, o));
  if (lane == 0) red[wid] = mx;
  __syncthreads();
  if (tid == 0) {
    float m2 = red[0];
#pragma unroll
    for (int i = 1; i < 8; ++i) m2 = fmaxf(m2, red[i]);
    red[15] = m2;
  }
  __syncthreads();
  const float M = red[15];

  float sum = 0.0f;
  for (int i = tid; i < V_; i += 256) {
    float e = expf(sh[i] - M);
    sh[i] = e;
    sum += e;
  }
#pragma unroll
  for (int o = 16; o > 0; o >>= 1) sum += __shfl_down(sum, o);
  if (lane == 0) red[wid] = sum;
  __syncthreads();
  if (tid == 0) {
    float z = 0.0f;
#pragma unroll
    for (int i = 0; i < 8; ++i) z += red[i];
    red[14] = 1.0f / z;
  }
  __syncthreads();
  const float inv = red[14];

  for (int i = tid; i < V_; i += 256) {
    float p = sh[i] * inv;
    if (w_bf) w_bf[(size_t)row * V_ + i] = (bf16_t)p;
    if (out)  out[((size_t)row * T_ + t) * V_ + i] = p;
  }
}

// ---------------- soft_emb = probs(BxV) @ emb(VxE) via embT -----------------
// grid: 16 blocks x 256 thr = 128 waves = 8 Mtiles x 16 Ntiles; K = 8192.

__global__ __launch_bounds__(256) void softemb_kernel(
    const bf16_t* __restrict__ w_bf,   // B x V bf16 probs
    const bf16_t* __restrict__ embT,   // E x V bf16
    bf16_t* __restrict__ x_bf) {       // B x E bf16
  const int lane = threadIdx.x & 31;
  const int wave = blockIdx.x * 8 + (threadIdx.x >> 5);
  const int mt = wave >> 4;       // 0..7
  const int nt = wave & 15;       // 0..15
  const int m0 = mt * 16;
  const int n0 = nt * 16;

  v8f acc = vzero8();
  for (int k0 = 0; k0 < V_; k0 += 32) {
    v16bf a  = load_a_frag(w_bf, V_, m0, k0, lane);
    v16bf bm = load_b_frag(embT, V_, n0, k0, lane);
    acc = wmma_bf16(a, bm, acc);
  }
  const int n  = n0 + (lane & 15);
  const int mb = m0 + 8 * (lane >> 4);
#pragma unroll
  for (int r = 0; r < 8; ++r)
    x_bf[(mb + r) * E_ + n] = (bf16_t)acc[r];
}

// ---------------------------- host orchestration ----------------------------

extern "C" void kernel_launch(void* const* d_in, const int* in_sizes, int n_in,
                              void* d_out, int out_size, void* d_ws,
                              size_t ws_size, hipStream_t stream) {
  const float* noise  = (const float*)d_in[0];
  const float* tau    = (const float*)d_in[1];
  const float* W_init = (const float*)d_in[2];
  const float* b_init = (const float*)d_in[3];
  const float* emb    = (const float*)d_in[4];
  const float* W_ih   = (const float*)d_in[5];
  const float* W_hh   = (const float*)d_in[6];
  const float* b_ih   = (const float*)d_in[7];
  const float* b_hh   = (const float*)d_in[8];
  const float* W_out  = (const float*)d_in[9];
  const float* b_out  = (const float*)d_in[10];
  float* out = (float*)d_out;

  // workspace carve-out (256B aligned), ~39 MB total
  char* ws = (char*)d_ws;
  size_t off = 0;
  auto carve = [&](size_t bytes) -> char* {
    char* p = ws + off;
    off = (off + bytes + 255) & ~(size_t)255;
    return p;
  };
  bf16_t* Whh_bf  = (bf16_t*)carve((size_t)H4_ * H_ * 2);
  bf16_t* Wih_bf  = (bf16_t*)carve((size_t)H4_ * E_ * 2);
  bf16_t* Wout_bf = (bf16_t*)carve((size_t)V_ * H_ * 2);
  bf16_t* embT    = (bf16_t*)carve((size_t)E_ * V_ * 2);
  float*  bias    = (float*) carve((size_t)H4_ * 4);
  bf16_t* hA      = (bf16_t*)carve((size_t)B_ * H_ * 2);
  bf16_t* hB      = (bf16_t*)carve((size_t)B_ * H_ * 2);
  float*  cbuf    = (float*) carve((size_t)B_ * H_ * 4);
  float*  sbuf    = (float*) carve((size_t)B_ * V_ * 4);
  bf16_t* wprob   = (bf16_t*)carve((size_t)B_ * V_ * 2);
  bf16_t* xemb    = (bf16_t*)carve((size_t)B_ * E_ * 2);
  (void)ws_size; (void)in_sizes; (void)n_in; (void)out_size;

  // one-time (per call) weight conversion — deterministic, replay-safe
  {
    int n;
    n = H4_ * H_;
    cvt_bf16_kernel<<<(n + 255) / 256, 256, 0, stream>>>(W_hh, Whh_bf, n);
    n = H4_ * E_;
    cvt_bf16_kernel<<<(n + 255) / 256, 256, 0, stream>>>(W_ih, Wih_bf, n);
    n = V_ * H_;
    cvt_bf16_kernel<<<(n + 255) / 256, 256, 0, stream>>>(W_out, Wout_bf, n);
    n = V_ * E_;
    embT_kernel<<<(n + 255) / 256, 256, 0, stream>>>(emb, embT);
    bias_kernel<<<(H4_ + 255) / 256, 256, 0, stream>>>(b_ih, b_hh, bias);
    n = B_ * 2 * H_;
    init_state_kernel<<<(n + 255) / 256, 256, 0, stream>>>(noise, W_init,
                                                           b_init, hA, cbuf);
  }

  bf16_t* h_cur = hA;
  bf16_t* h_nxt = hB;

  for (int t = 0; t < T_; ++t) {
    // cell_a: x = pad_emb == emb[0] == 0  -> x term vanishes
    lstm_cell_kernel<<<32, 256, 0, stream>>>(h_cur, nullptr, cbuf, h_nxt,
                                             Whh_bf, Wih_bf, bias);
    { bf16_t* tmp = h_cur; h_cur = h_nxt; h_nxt = tmp; }

    logits_gumbel_kernel<<<64, 256, 0, stream>>>(
        h_cur, Wout_bf, b_out, tau, sbuf, (unsigned)(2 * t + 1));
    softmax_kernel<<<128, 256, 0, stream>>>(sbuf, wprob, nullptr, t);
    softemb_kernel<<<16, 256, 0, stream>>>(wprob, embT, xemb);

    // cell_b: x = (t==0) ? pad_emb(=0) : soft_emb
    lstm_cell_kernel<<<32, 256, 0, stream>>>(
        h_cur, (t == 0) ? nullptr : xemb, cbuf, h_nxt, Whh_bf, Wih_bf, bias);
    { bf16_t* tmp = h_cur; h_cur = h_nxt; h_nxt = tmp; }

    logits_gumbel_kernel<<<64, 256, 0, stream>>>(
        h_cur, Wout_bf, b_out, tau, sbuf, (unsigned)(2 * t + 2));
    softmax_kernel<<<128, 256, 0, stream>>>(sbuf, nullptr, out, t);
  }
}
